// DeformableTransformer_30116310679588
// MI455X (gfx1250) — compile-verified, matrix operand order checked by reading, not run on hardware
//
#include <hip/hip_runtime.h>
#include <hip/hip_bf16.h>
#include <math.h>

// ---- problem constants (match reference) ----
#define BB   2
#define DDIM 256
#define DFFN 1024
#define NHH  8
#define NQQ  300
#define LENN 5440   // 64*64 + 32*32 + 16*16 + 8*8

typedef __attribute__((ext_vector_type(16))) __bf16 bf16x16;
typedef __attribute__((ext_vector_type(8)))  float  f32x8;

// ---------------------------------------------------------------------------
// Pack f32 weight W[K,N] into bf16 WMMA B-fragment layout:
//   P[((kt*(N/16) + nt)*32 + lane)*16 + j]
// lane<16: col = nt*16+lane, k = kt*32 + {j<8 ? j : 16+(j-8)}
// lane>=16: same col, k += 8.  One thread per packed element.
// ---------------------------------------------------------------------------
__global__ void pack_w_kernel(const float* __restrict__ W,
                              __bf16* __restrict__ P, int N, int K)
{
  int t = blockIdx.x * 256 + threadIdx.x;
  if (t >= K * N) return;
  int j    = t & 15;
  int lane = (t >> 4) & 31;
  int rest = t >> 9;
  int ntiles = N >> 4;
  int nt = rest % ntiles;
  int kt = rest / ntiles;
  int hl = lane >> 4, ll = lane & 15;
  int k = kt * 32 + hl * 8 + (j < 8 ? j : 16 + (j - 8));
  int n = nt * 16 + ll;
  P[t] = (__bf16)W[(size_t)k * N + n];
}

// ---------------------------------------------------------------------------
// GEMM: C[M,N] = act(A[M,K] @ W[K,N] + bias); f32 A/C, pre-packed bf16 W.
// Block = 256 threads (8 waves). Each wave owns a 64x16 C strip: 4 M-tiles
// sharing one B fragment per k-step -> 4 v_wmma per iteration.
// Ragged M handled by clamping A-row (load-safe) and guarding stores only.
// ---------------------------------------------------------------------------
__global__ void __launch_bounds__(256)
gemm_bf16_wmma(const float* __restrict__ A, const __bf16* __restrict__ WP,
               const float* __restrict__ bias, float* __restrict__ C,
               int M, int N, int K, int relu)
{
  const int lane = threadIdx.x & 31;
  const int wave = threadIdx.x >> 5;
  const int nt   = blockIdx.x * 8 + wave;       // 16-col tile index
  const int tn   = nt * 16;
  const int tm   = blockIdx.y * 64;
  const int hl   = lane >> 4;                   // lane half: 0/1
  const int ll   = lane & 15;
  const int kgrp = hl * 8;
  const int ntiles = N >> 4;

  // A row pointers (clamped: extra rows recompute row M-1, stores guarded)
  const float* Ap[4];
#pragma unroll
  for (int r = 0; r < 4; ++r) {
    int ar = tm + r * 16 + ll;
    ar = ar < M ? ar : (M - 1);
    Ap[r] = A + (size_t)ar * K + kgrp;
  }
  const __bf16* Bp = WP + ((size_t)nt * 32 + lane) * 16;
  const size_t bstride = (size_t)ntiles * 512;  // packed elems per k-tile

  f32x8 acc[4] = {{}, {}, {}, {}};
  for (int kt = 0; kt < (K >> 5); ++kt) {
    bf16x16 b = *reinterpret_cast<const bf16x16*>(Bp + (size_t)kt * bstride);
    __builtin_prefetch(Bp + (size_t)(kt + 1) * bstride, 0, 0);
#pragma unroll
    for (int r = 0; r < 4; ++r) {
      const float* ap = Ap[r] + kt * 32;
      float4 v0 = *(const float4*)(ap);
      float4 v1 = *(const float4*)(ap + 4);
      float4 v2 = *(const float4*)(ap + 16);
      float4 v3 = *(const float4*)(ap + 20);
      bf16x16 a;
      a[0]  = (__bf16)v0.x; a[1]  = (__bf16)v0.y; a[2]  = (__bf16)v0.z; a[3]  = (__bf16)v0.w;
      a[4]  = (__bf16)v1.x; a[5]  = (__bf16)v1.y; a[6]  = (__bf16)v1.z; a[7]  = (__bf16)v1.w;
      a[8]  = (__bf16)v2.x; a[9]  = (__bf16)v2.y; a[10] = (__bf16)v2.z; a[11] = (__bf16)v2.w;
      a[12] = (__bf16)v3.x; a[13] = (__bf16)v3.y; a[14] = (__bf16)v3.z; a[15] = (__bf16)v3.w;
      acc[r] = __builtin_amdgcn_wmma_f32_16x16x32_bf16(false, a, false, b,
                                                       (short)0, acc[r], false, false);
    }
  }

  const int cb = tn + ll;
  const float bv = bias ? bias[cb] : 0.f;
#pragma unroll
  for (int r = 0; r < 4; ++r) {
#pragma unroll
    for (int v = 0; v < 8; ++v) {
      int rr = tm + r * 16 + v + hl * 8;
      if (rr < M) {
        float x = acc[r][v] + bv;
        if (relu) x = fmaxf(x, 0.f);
        C[(size_t)rr * N + cb] = x;
      }
    }
  }
}

// ---------------------------------------------------------------------------
// y = a + b (elementwise)
// ---------------------------------------------------------------------------
__global__ void add_kernel(const float* __restrict__ a,
                           const float* __restrict__ b,
                           float* __restrict__ y, int n)
{
  int t = blockIdx.x * 256 + threadIdx.x;
  if (t < n) y[t] = a[t] + b[t];
}

// ---------------------------------------------------------------------------
// y = LayerNorm(x + r) * g + b ; one 256-thread block per 256-wide row.
// Safe in-place (block reads its whole row before writing).
// ---------------------------------------------------------------------------
__global__ void __launch_bounds__(256)
add_ln_kernel(const float* __restrict__ x, const float* __restrict__ r,
              float* __restrict__ y, const float* __restrict__ g,
              const float* __restrict__ b)
{
  __shared__ float sm[256];
  const int row = blockIdx.x;
  const int t = threadIdx.x;
  float v = x[(size_t)row * 256 + t] + r[(size_t)row * 256 + t];
  sm[t] = v; __syncthreads();
  for (int s = 128; s > 0; s >>= 1) { if (t < s) sm[t] += sm[t + s]; __syncthreads(); }
  float mean = sm[0] * (1.f / 256.f);
  __syncthreads();
  float d = v - mean;
  sm[t] = d * d; __syncthreads();
  for (int s = 128; s > 0; s >>= 1) { if (t < s) sm[t] += sm[t + s]; __syncthreads(); }
  float var = sm[0] * (1.f / 256.f);
  y[(size_t)row * 256 + t] = d * rsqrtf(var + 1e-5f) * g[t] + b[t];
}

// ---------------------------------------------------------------------------
// softmax over groups of 16 (per head, NL*NP=16) within rows of 128.
// One thread per (row, head). In-place safe.
// ---------------------------------------------------------------------------
__global__ void softmax16_kernel(float* __restrict__ w, int rows)
{
  int t = blockIdx.x * 256 + threadIdx.x;
  if (t >= rows * 8) return;
  float* p = w + (size_t)(t >> 3) * 128 + (t & 7) * 16;
  float m = p[0];
#pragma unroll
  for (int j = 1; j < 16; ++j) m = fmaxf(m, p[j]);
  float e[16], s = 0.f;
#pragma unroll
  for (int j = 0; j < 16; ++j) { e[j] = __expf(p[j] - m); s += e[j]; }
  float inv = 1.f / s;
#pragma unroll
  for (int j = 0; j < 16; ++j) p[j] = e[j] * inv;
}

// ---------------------------------------------------------------------------
// Multi-scale deformable attention sampling + head-weighted sum.
// Thread per output channel (b, i, c), c = h*32 + dh. 4 levels x 4 points,
// bilinear gather from val (B, LEN, 256). refbuf==nullptr -> encoder refs
// computed from the query's own cell center.
// ---------------------------------------------------------------------------
__global__ void deform_attn_kernel(const float* __restrict__ val,
                                   const float* __restrict__ off,
                                   const float* __restrict__ wgt,
                                   const float* __restrict__ refbuf,
                                   float* __restrict__ out, int Lq)
{
  const int LST[4] = {0, 4096, 5120, 5376};
  const int LH [4] = {64, 32, 16, 8};
  int tid = blockIdx.x * 256 + threadIdx.x;
  int total = BB * Lq * DDIM;
  if (tid >= total) return;
  int c = tid & 255;
  int i = (tid >> 8) % Lq;
  int b = tid / (Lq * DDIM);
  int h = c >> 5;

  float rx, ry;
  if (refbuf) {
    rx = refbuf[((size_t)b * Lq + i) * 2 + 0];
    ry = refbuf[((size_t)b * Lq + i) * 2 + 1];
  } else {
    int lvl, j;
    if (i < 4096)      { lvl = 0; j = i; }
    else if (i < 5120) { lvl = 1; j = i - 4096; }
    else if (i < 5376) { lvl = 2; j = i - 5120; }
    else               { lvl = 3; j = i - 5376; }
    int Wl = LH[lvl];
    int iy = j / Wl, ix = j % Wl;
    rx = (ix + 0.5f) / (float)Wl;
    ry = (iy + 0.5f) / (float)Wl;
  }

  const float* offrow = off + ((size_t)b * Lq + i) * 256;
  const float* wrow   = wgt + ((size_t)b * Lq + i) * 128;
  float accv = 0.f;
#pragma unroll
  for (int l = 0; l < 4; ++l) {
    const int H_ = LH[l], W_ = LH[l], st = LST[l];
    const float* vb = val + ((size_t)b * LENN + st) * 256 + c;
#pragma unroll
    for (int p = 0; p < 4; ++p) {
      int oc = ((h * 4 + l) * 4 + p) * 2;
      float gx = rx + offrow[oc]     / (float)W_;   // NORM_WH = (W, H)
      float gy = ry + offrow[oc + 1] / (float)H_;
      float wt = wrow[h * 16 + l * 4 + p];
      float x = gx * (float)W_ - 0.5f;
      float y = gy * (float)H_ - 0.5f;
      float x0f = floorf(x), y0f = floorf(y);
      int x0 = (int)x0f, y0 = (int)y0f;
      float wx = x - x0f, wy = y - y0f;
      float s = 0.f;
#pragma unroll
      for (int dyi = 0; dyi < 2; ++dyi)
#pragma unroll
        for (int dxi = 0; dxi < 2; ++dxi) {
          int xi = x0 + dxi, yi = y0 + dyi;
          if (xi >= 0 && xi < W_ && yi >= 0 && yi < H_) {
            float cw = (dxi ? wx : 1.f - wx) * (dyi ? wy : 1.f - wy);
            s += cw * vb[(size_t)(yi * W_ + xi) * 256];
          }
        }
      accv += wt * s;
    }
  }
  out[((size_t)b * Lq + i) * 256 + c] = accv;
}

// ---------------------------------------------------------------------------
// MHA with online softmax: one wave32 per (b, h, q); lane = head dim (DH=32).
// Scores via __shfl_xor wave reduction (wave32-correct).
// ---------------------------------------------------------------------------
__global__ void __launch_bounds__(256)
mha_kernel(const float* __restrict__ qh, const float* __restrict__ kh,
           const float* __restrict__ vh, float* __restrict__ o, int L)
{
  int wid  = blockIdx.x * 8 + (threadIdx.x >> 5);
  int lane = threadIdx.x & 31;
  int total = BB * NHH * L;
  if (wid >= total) return;
  int qi = wid % L;
  int h  = (wid / L) % NHH;
  int b  = wid / (L * NHH);
  const float scale = 0.17677669529663687f;   // 32^-0.5
  float qv = qh[((size_t)b * L + qi) * 256 + h * 32 + lane] * scale;
  float m = -3.4e38f, s = 0.f, acc = 0.f;
  for (int k = 0; k < L; ++k) {
    float d = qv * kh[((size_t)b * L + k) * 256 + h * 32 + lane];
#pragma unroll
    for (int off = 16; off >= 1; off >>= 1) d += __shfl_xor(d, off, 32);
    float mn   = fmaxf(m, d);
    float corr = __expf(m - mn);
    float p    = __expf(d - mn);
    float vv   = vh[((size_t)b * L + k) * 256 + h * 32 + lane];
    s   = s * corr + p;
    acc = acc * corr + p * vv;
    m = mn;
  }
  o[((size_t)b * L + qi) * 256 + h * 32 + lane] = acc / s;
}

// ---------------------------------------------------------------------------
// Decoder init: tgt = query_embed[:, D:], qp = query_embed[:, :D] (bcast B)
// ---------------------------------------------------------------------------
__global__ void init_dec_kernel(const float* __restrict__ qe,
                                float* __restrict__ tgt, float* __restrict__ qp)
{
  int t = blockIdx.x * 256 + threadIdx.x;
  int total = BB * NQQ * 256;
  if (t >= total) return;
  int c = t & 255;
  int i = (t >> 8) % NQQ;
  tgt[t] = qe[(size_t)i * 512 + 256 + c];
  qp [t] = qe[(size_t)i * 512 + c];
}

// ---------------------------------------------------------------------------
// ref = sigmoid(qp @ Wref + bref); thread per (b, i, j) with j in {0,1}
// ---------------------------------------------------------------------------
__global__ void ref_kernel(const float* __restrict__ qe,
                           const float* __restrict__ Wref,
                           const float* __restrict__ bref,
                           float* __restrict__ ref)
{
  int t = blockIdx.x * 256 + threadIdx.x;
  if (t >= BB * NQQ * 2) return;
  int j = t & 1;
  int i = (t >> 1) % NQQ;
  float acc = bref[j];
  for (int k = 0; k < 256; ++k) acc += qe[(size_t)i * 512 + k] * Wref[k * 2 + j];
  ref[t] = 1.f / (1.f + __expf(-acc));
}

// ===========================================================================
extern "C" void kernel_launch(void* const* d_in, const int* in_sizes, int n_in,
                              void* d_out, int out_size, void* d_ws, size_t ws_size,
                              hipStream_t stream)
{
  (void)in_sizes; (void)n_in; (void)out_size; (void)ws_size;
  const float* src = (const float*)d_in[0];
  const float* pos = (const float*)d_in[1];
  const float* qe  = (const float*)d_in[2];
  auto enc = [&](int l, int i) { return (const float*)d_in[3 + l * 16 + i]; };
  auto dec = [&](int l, int i) { return (const float*)d_in[3 + 48 + l * 26 + i]; };
  const float* Wref = (const float*)d_in[3 + 48 + 78 + 0];
  const float* bref = (const float*)d_in[3 + 48 + 78 + 1];

  float* ws = (float*)d_ws;
  const size_t ME = (size_t)BB * LENN;   // 10880 encoder rows (170 * 64)
  const size_t MD = (size_t)BB * NQQ;    // 600 decoder rows
  size_t off = 0;
  float* X    = ws + off; off += ME * 256;
  float* T1   = ws + off; off += ME * 256;
  float* T2   = ws + off; off += ME * 256;
  float* T3   = ws + off; off += ME * 256;   // val (enc + dec memory-val)
  float* T4   = ws + off; off += ME * 256;
  float* WB   = ws + off; off += ME * 128;
  float* FF   = ws + off; off += ME * 1024;
  float* DTGT = ws + off; off += MD * 256;
  float* DQP  = ws + off; off += MD * 256;
  float* DQ   = ws + off; off += MD * 256;
  float* DQH  = ws + off; off += MD * 256;
  float* DKH  = ws + off; off += MD * 256;
  float* DVH  = ws + off; off += MD * 256;
  float* DSA  = ws + off; off += MD * 256;
  float* DOFF = ws + off; off += MD * 256;
  float* DW   = ws + off; off += MD * 128;
  float* DREF = ws + off; off += MD * 2;     // 1200 floats (keeps 32B align)
  float* DFF  = ws + off; off += MD * 1024;
  float* DTMP = ws + off; off += MD * 256;
  __bf16* PW  = (__bf16*)(ws + off); off += (size_t)DFFN * DDIM / 2; // 256K bf16

  auto gemm = [&](const float* A, const float* Wm, const float* bias, float* C,
                  int M, int N, int K, int relu) {
    pack_w_kernel<<<(unsigned)(((size_t)K * N + 255) / 256), 256, 0, stream>>>(
        Wm, PW, N, K);
    dim3 g(N / 128, (M + 63) / 64);   // N in {128,256,1024}
    gemm_bf16_wmma<<<g, 256, 0, stream>>>(A, PW, bias, C, M, N, K, relu);
  };
  auto add = [&](const float* a, const float* b2, float* y, size_t n) {
    add_kernel<<<(unsigned)((n + 255) / 256), 256, 0, stream>>>(a, b2, y, (int)n);
  };
  auto addln = [&](const float* x, const float* r, float* y,
                   const float* g, const float* b2, int M) {
    add_ln_kernel<<<M, 256, 0, stream>>>(x, r, y, g, b2);
  };

  // ---------------- encoder ----------------
  const float* cur = src;
  for (int l = 0; l < 3; ++l) {
    add(cur, pos, T1, ME * 256);                                   // q = x + pos
    gemm(T1, enc(l, 0), enc(l, 1), T2, (int)ME, 256, 256, 0);      // off
    gemm(T1, enc(l, 2), enc(l, 3), WB, (int)ME, 128, 256, 0);      // w logits
    softmax16_kernel<<<(unsigned)((ME * 8 + 255) / 256), 256, 0, stream>>>(WB, (int)ME);
    gemm(cur, enc(l, 4), enc(l, 5), T3, (int)ME, 256, 256, 0);     // val
    deform_attn_kernel<<<(unsigned)((ME * 256 + 255) / 256), 256, 0, stream>>>(
        T3, T2, WB, nullptr, T4, LENN);
    gemm(T4, enc(l, 6), enc(l, 7), T1, (int)ME, 256, 256, 0);      // a = attn@Wo
    addln(cur, T1, X, enc(l, 8), enc(l, 9), (int)ME);              // LN1
    gemm(X,  enc(l, 10), enc(l, 11), FF, (int)ME, 1024, 256, 1);   // relu FFN1
    gemm(FF, enc(l, 12), enc(l, 13), T4, (int)ME, 256, 1024, 0);   // FFN2
    addln(X, T4, X, enc(l, 14), enc(l, 15), (int)ME);              // LN2
    cur = X;
  }
  const float* mem = X;

  // ---------------- decoder ----------------
  init_dec_kernel<<<(unsigned)((MD * 256 + 255) / 256), 256, 0, stream>>>(qe, DTGT, DQP);
  ref_kernel<<<(unsigned)((MD * 2 + 255) / 256), 256, 0, stream>>>(qe, Wref, bref, DREF);

  for (int l = 0; l < 3; ++l) {
    // self-attention
    add(DTGT, DQP, DQ, MD * 256);                                  // q = tgt + qp
    gemm(DQ,   dec(l, 16), dec(l, 17), DQH, (int)MD, 256, 256, 0); // Wq
    gemm(DQ,   dec(l, 18), dec(l, 19), DKH, (int)MD, 256, 256, 0); // Wk
    gemm(DTGT, dec(l, 20), dec(l, 21), DVH, (int)MD, 256, 256, 0); // Wv2
    mha_kernel<<<(unsigned)((BB * NHH * NQQ + 7) / 8), 256, 0, stream>>>(
        DQH, DKH, DVH, DTMP, NQQ);
    gemm(DTMP, dec(l, 22), dec(l, 23), DSA, (int)MD, 256, 256, 0); // Wao
    addln(DTGT, DSA, DTGT, dec(l, 14), dec(l, 15), (int)MD);       // LN (g2,b2)
    // cross deformable attention
    add(DTGT, DQP, DQ, MD * 256);
    gemm(DQ, dec(l, 0), dec(l, 1), DOFF, (int)MD, 256, 256, 0);    // off
    gemm(DQ, dec(l, 2), dec(l, 3), DW,   (int)MD, 128, 256, 0);    // w logits
    softmax16_kernel<<<(unsigned)((MD * 8 + 255) / 256), 256, 0, stream>>>(DW, (int)MD);
    gemm(mem, dec(l, 4), dec(l, 5), T3, (int)ME, 256, 256, 0);     // val(memory)
    deform_attn_kernel<<<(unsigned)((MD * 256 + 255) / 256), 256, 0, stream>>>(
        T3, DOFF, DW, DREF, DTMP, NQQ);
    gemm(DTMP, dec(l, 6), dec(l, 7), DSA, (int)MD, 256, 256, 0);   // Wo
    addln(DTGT, DSA, DTGT, dec(l, 8), dec(l, 9), (int)MD);         // LN (g1,b1)
    // FFN
    gemm(DTGT, dec(l, 10), dec(l, 11), DFF, (int)MD, 1024, 256, 1);
    gemm(DFF,  dec(l, 12), dec(l, 13), DTMP, (int)MD, 256, 1024, 0);
    addln(DTGT, DTMP, DTGT, dec(l, 24), dec(l, 25), (int)MD);      // LN (g3,b3)
  }

  hipMemcpyAsync(d_out, DTGT, MD * 256 * sizeof(float),
                 hipMemcpyDeviceToDevice, stream);
}
